// HierarchicalMultiHeadAttention_29557964931133
// MI455X (gfx1250) — matrix operand order repaired and checked by reading, not compiled, not run on hardware
//
#include <hip/hip_runtime.h>

#define DMODEL 1024
#define NHEAD  16
#define HDIM   64
#define GEMM_NK (DMODEL / 32)

typedef __attribute__((ext_vector_type(16))) __bf16         bf16x16;
typedef __attribute__((ext_vector_type(8)))  unsigned short u16x8;
typedef __attribute__((ext_vector_type(4)))  unsigned short u16x4;
typedef __attribute__((ext_vector_type(8)))  float          f32x8;
typedef __attribute__((ext_vector_type(4)))  unsigned int   u32x4;
typedef __attribute__((ext_vector_type(8)))  unsigned int   u32x8;
typedef unsigned short ushort;

union Frag {
    bf16x16 v;
    u16x8   h[2];
    unsigned short u[16];
};

__device__ __forceinline__ ushort f2bf(float f) {
    unsigned u = __float_as_uint(f);
    u += 0x7FFFu + ((u >> 16) & 1u);      // round to nearest even
    return (ushort)(u >> 16);
}

__device__ __forceinline__ f32x8 wmma_bf16(const Frag& a, const Frag& b, f32x8 c) {
    return __builtin_amdgcn_wmma_f32_16x16x32_bf16(
        false, a.v, false, b.v, (short)0, c, false, false);
}

// LDS byte offset of a __shared__ object (generic -> addrspace(3) -> int)
__device__ __forceinline__ unsigned lds_off(const void* p) {
    return (unsigned)(unsigned long long)(const __attribute__((address_space(3))) char*)p;
}

// ---- Tensor Data Mover: 2D tile (bf16) global -> LDS, tracked by TENSORcnt.
__device__ __forceinline__ void tdm_load_tile_2d(const void* gptr,
                                                 unsigned lds_byte_off,
                                                 unsigned tile_k,      // dim0 (contiguous) elems
                                                 unsigned tile_rows,   // dim1
                                                 unsigned row_stride_elems) {
    unsigned long long ga = (unsigned long long)(size_t)gptr;
    u32x4 g0;
    g0[0] = 1u;                                           // count=1, user mode
    g0[1] = lds_byte_off;                                 // lds_addr
    g0[2] = (unsigned)ga;                                 // global_addr[31:0]
    g0[3] = ((unsigned)(ga >> 32) & 0x1FFFFFFu) | (2u << 30);  // addr[56:32], type=2
    u32x8 g1;
    g1[0] = 1u << 16;                                     // data_size=1 (2 bytes)
    g1[1] = (tile_k & 0xFFFFu) << 16;                     // tensor_dim0[15:0]
    g1[2] = (tile_k >> 16) | ((tile_rows & 0xFFFFu) << 16);
    g1[3] = (tile_rows >> 16) | (tile_k << 16);           // dim1 hi | tile_dim0
    g1[4] = tile_rows;                                    // tile_dim1 (tile_dim2=0)
    g1[5] = row_stride_elems;                             // tensor_dim0_stride[31:0]
    g1[6] = 0u;
    g1[7] = 0u;
    asm volatile("tensor_load_to_lds %0, %1" :: "s"(g0), "s"(g1) : "memory");
}

// ---------------------------------------------------------------- converts
__global__ void k_f32_to_bf16(const float* __restrict__ x, ushort* __restrict__ y, int n4) {
    int i = blockIdx.x * blockDim.x + threadIdx.x;
    if (i >= n4) return;
    float4 f = reinterpret_cast<const float4*>(x)[i];
    u16x4 o;
    o.x = f2bf(f.x); o.y = f2bf(f.y); o.z = f2bf(f.z); o.w = f2bf(f.w);
    reinterpret_cast<u16x4*>(y)[i] = o;
}

// W[k][n] f32 -> WT[n][k] bf16 (so WMMA B-fragments are contiguous loads)
__global__ void k_transpose_w(const float* __restrict__ w, ushort* __restrict__ wt) {
    __shared__ float tile[32][33];
    int k0 = blockIdx.x * 32, n0 = blockIdx.y * 32;
    for (int r = threadIdx.y; r < 32; r += 8)
        tile[r][threadIdx.x] = w[(size_t)(k0 + r) * DMODEL + n0 + threadIdx.x];
    __syncthreads();
    for (int r = threadIdx.y; r < 32; r += 8)
        wt[(size_t)(n0 + r) * DMODEL + k0 + threadIdx.x] = f2bf(tile[threadIdx.x][r]);
}

// ---------------------------------------------------------------- GEMM
// One wave computes a 64x64 tile of  out = (A[M,1024] @ W + bias) * scale,
// with TDM double-buffered LDS staging of A and WT k-slabs.
// mode 0: bf16 [M,D]   mode 1: f32 [M,D]   mode 2: bf16 head-transposed VT
__global__ void __launch_bounds__(32)
k_gemm_bf16(const ushort* __restrict__ A, const ushort* __restrict__ WT,
            const float* __restrict__ bias, ushort* __restrict__ outB,
            float* __restrict__ outF, int Llog2, float scale, int mode) {
    constexpr int TILE = 64 * 32;                 // elements per k-slab
    __shared__ __align__(16) ushort lA[2][TILE];
    __shared__ __align__(16) ushort lB[2][TILE];
    const int lane = threadIdx.x & 31;
    const int half = lane >> 4;
    const int lr   = lane & 15;
    const int m0   = blockIdx.x * 64;
    const int n0   = blockIdx.y * 64;

    const ushort* Abase = A  + (size_t)m0 * DMODEL;
    const ushort* Bbase = WT + (size_t)n0 * DMODEL;
    const unsigned offA0 = lds_off(lA[0]);
    const unsigned offB0 = lds_off(lB[0]);

    // prologue: DMA k-step 0 into buffer 0
    tdm_load_tile_2d(Abase, offA0, 32, 64, DMODEL);
    tdm_load_tile_2d(Bbase, offB0, 32, 64, DMODEL);

    f32x8 acc[4][4] = {};

#pragma unroll 1
    for (int ks = 0; ks < GEMM_NK; ++ks) {
        const int cur = ks & 1;
        if (ks + 1 < GEMM_NK) {
            // buffer cur^1 was consumed in iter ks-1; make sure LDS reads drained
            asm volatile("s_wait_dscnt 0x0" ::: "memory");
            const unsigned nb = (unsigned)(cur ^ 1) * (TILE * 2u);
            tdm_load_tile_2d(Abase + (size_t)(ks + 1) * 32, offA0 + nb, 32, 64, DMODEL);
            tdm_load_tile_2d(Bbase + (size_t)(ks + 1) * 32, offB0 + nb, 32, 64, DMODEL);
            __builtin_amdgcn_s_wait_tensorcnt(2);   // current pair done; next in flight
        } else {
            __builtin_amdgcn_s_wait_tensorcnt(0);
        }
        const ushort* a_t = lA[0] + cur * TILE;
        const ushort* b_t = lB[0] + cur * TILE;
        Frag af[4];
#pragma unroll
        for (int mt = 0; mt < 4; ++mt) {
            const ushort* ar = a_t + (mt * 16 + lr) * 32 + half * 8;
            af[mt].h[0] = *reinterpret_cast<const u16x8*>(ar);
            af[mt].h[1] = *reinterpret_cast<const u16x8*>(ar + 16);
        }
        // B fragments software-pipelined: load nt+1 before consuming nt
        Frag bf[2];
        auto loadB = [&](int nt, int buf) {
            const ushort* br = b_t + (nt * 16 + lr) * 32 + half * 8;
            bf[buf].h[0] = *reinterpret_cast<const u16x8*>(br);
            bf[buf].h[1] = *reinterpret_cast<const u16x8*>(br + 16);
        };
        loadB(0, 0);
#pragma unroll
        for (int nt = 0; nt < 4; ++nt) {
            if (nt + 1 < 4) loadB(nt + 1, (nt + 1) & 1);
#pragma unroll
            for (int mt = 0; mt < 4; ++mt)
                acc[mt][nt] = wmma_bf16(af[mt], bf[nt & 1], acc[mt][nt]);
        }
    }

    const int Lm1 = (1 << Llog2) - 1;
#pragma unroll
    for (int nt = 0; nt < 4; ++nt) {
        const int col = n0 + nt * 16 + lr;
        const float bv = bias[col];
#pragma unroll
        for (int mt = 0; mt < 4; ++mt) {
#pragma unroll
            for (int e = 0; e < 8; ++e) {
                int row = m0 + mt * 16 + e + 8 * half;
                float val = (acc[mt][nt][e] + bv) * scale;
                if (mode == 0) {
                    outB[(size_t)row * DMODEL + col] = f2bf(val);
                } else if (mode == 1) {
                    outF[(size_t)row * DMODEL + col] = val;
                } else {
                    int bb = row >> Llog2, ll = row & Lm1;
                    int hh = col >> 6, dd = col & 63;
                    outB[((size_t)((bb * NHEAD + hh) * HDIM + dd) << Llog2) + ll] = f2bf(val);
                }
            }
        }
    }
}

// ---------------------------------------------------------------- attention
template <int L>
__global__ void __launch_bounds__(32)
k_attn(const ushort* __restrict__ Q, const ushort* __restrict__ Kp,
       const ushort* __restrict__ VT, const float* __restrict__ mask,
       float* __restrict__ attn_out, ushort* __restrict__ ctx_out) {
    extern __shared__ float sc[];                 // [16][L+16]
    constexpr int SLP = L + 16;
    const int lane = threadIdx.x & 31;
    const int half = lane >> 4;
    const int lr   = lane & 15;
    const int qt = blockIdx.x, h = blockIdx.y, b = blockIdx.z;
    const int qbase = qt * 16;

    const ushort* qrow = Q + (size_t)(b * L + qbase + lr) * DMODEL + h * HDIM;
    Frag a0, a1;                                  // Q tile 16x64 (1/32 folded in)
    a0.h[0] = *reinterpret_cast<const u16x8*>(qrow + half * 8);
    a0.h[1] = *reinterpret_cast<const u16x8*>(qrow + 16 + half * 8);
    a1.h[0] = *reinterpret_cast<const u16x8*>(qrow + 32 + half * 8);
    a1.h[1] = *reinterpret_cast<const u16x8*>(qrow + 48 + half * 8);

    const float* mbase = mask + (size_t)(b * NHEAD + h) * L * L;

    // ---- phase 1: masked scores -> LDS, K fragments double-buffered
    Frag kb0[2], kb1[2];
    auto loadK = [&](int kb, int buf) {
        const ushort* krow = Kp + (size_t)(b * L + kb * 16 + lr) * DMODEL + h * HDIM;
        kb0[buf].h[0] = *reinterpret_cast<const u16x8*>(krow + half * 8);
        kb0[buf].h[1] = *reinterpret_cast<const u16x8*>(krow + 16 + half * 8);
        kb1[buf].h[0] = *reinterpret_cast<const u16x8*>(krow + 32 + half * 8);
        kb1[buf].h[1] = *reinterpret_cast<const u16x8*>(krow + 48 + half * 8);
    };
    loadK(0, 0);
#pragma unroll 2
    for (int kb = 0; kb < L / 16; ++kb) {
        const int cur = kb & 1;
        if (kb + 1 < L / 16) loadK(kb + 1, cur ^ 1);
        f32x8 s = {};
        s = wmma_bf16(a0, kb0[cur], s);
        s = wmma_bf16(a1, kb1[cur], s);
#pragma unroll
        for (int e = 0; e < 8; ++e) {
            int mr  = e + 8 * half;
            int key = kb * 16 + lr;
            sc[mr * SLP + key] = s[e] * mbase[(size_t)(qbase + mr) * L + key];
        }
    }
    __syncthreads();

    // ---- phase 2: softmax (lane -> row lr, half-wave splits keys)
    constexpr int L2 = L / 2;
    float4* row4 = reinterpret_cast<float4*>(sc + lr * SLP + half * L2);
    float mx = -3.4e38f;
    for (int i = 0; i < L2 / 4; ++i) {
        float4 v = row4[i];
        mx = fmaxf(mx, fmaxf(fmaxf(v.x, v.y), fmaxf(v.z, v.w)));
    }
    mx = fmaxf(mx, __shfl_xor(mx, 16, 32));
    float sum = 0.f;
    for (int i = 0; i < L2 / 4; ++i) {
        float4 v = row4[i];
        v.x = __expf(v.x - mx); v.y = __expf(v.y - mx);
        v.z = __expf(v.z - mx); v.w = __expf(v.w - mx);
        row4[i] = v;
        sum += v.x + v.y + v.z + v.w;
    }
    sum += __shfl_xor(sum, 16, 32);
    float inv = 1.0f / sum;
    float4* aout4 = reinterpret_cast<float4*>(
        attn_out + (size_t)((b * NHEAD + h) * L + qbase + lr) * L + half * L2);
    for (int i = 0; i < L2 / 4; ++i) {
        float4 v = row4[i];
        v.x *= inv; v.y *= inv; v.z *= inv; v.w *= inv;
        row4[i] = v;
        aout4[i] = v;
    }
    __syncthreads();

    // ---- phase 3: ctx = P @ V, V fragments double-buffered
    f32x8 acc[4] = {};
    const float* prow = sc + lr * SLP;
    Frag vf[2][4];
    auto loadV = [&](int kb, int buf) {
#pragma unroll
        for (int t = 0; t < 4; ++t) {
            const ushort* vrow = VT +
                (size_t)((b * NHEAD + h) * HDIM + t * 16 + lr) * L + kb * 32 + half * 8;
            vf[buf][t].h[0] = *reinterpret_cast<const u16x8*>(vrow);
            vf[buf][t].h[1] = *reinterpret_cast<const u16x8*>(vrow + 16);
        }
    };
    loadV(0, 0);
#pragma unroll 2
    for (int kb = 0; kb < L / 32; ++kb) {
        const int cur = kb & 1;
        if (kb + 1 < L / 32) loadV(kb + 1, cur ^ 1);
        Frag a;
        const float* p0 = prow + kb * 32 + half * 8;
        const float* p1 = p0 + 16;
#pragma unroll
        for (int j = 0; j < 8; ++j) { a.u[j] = f2bf(p0[j]); a.u[8 + j] = f2bf(p1[j]); }
#pragma unroll
        for (int t = 0; t < 4; ++t)
            acc[t] = wmma_bf16(a, vf[cur][t], acc[t]);
    }
#pragma unroll
    for (int t = 0; t < 4; ++t) {
#pragma unroll
        for (int e = 0; e < 8; ++e) {
            int q   = qbase + e + 8 * half;
            int col = h * HDIM + t * 16 + lr;
            ctx_out[(size_t)(b * L + q) * DMODEL + col] = f2bf(acc[t][e]);
        }
    }
}

// ---------------------------------------------------------------- host
extern "C" void kernel_launch(void* const* d_in, const int* in_sizes, int n_in,
                              void* d_out, int out_size, void* d_ws, size_t ws_size,
                              hipStream_t stream) {
    (void)in_sizes; (void)n_in; (void)out_size; (void)ws_size;
    const int B = 2, SW = 1024, SS = 256;

    const float* word_inputs      = (const float*)d_in[0];
    const float* word_context     = (const float*)d_in[1];
    const float* sentence_inputs  = (const float*)d_in[2];
    const float* sentence_context = (const float*)d_in[3];
    const float* word_mask        = (const float*)d_in[4];
    const float* sentence_mask    = (const float*)d_in[5];

    float* out = (float*)d_out;
    const size_t word_out_off  = 0;
    const size_t sent_out_off  = (size_t)B * SW * DMODEL;
    const size_t word_attn_off = sent_out_off + (size_t)B * SS * DMODEL;
    const size_t sent_attn_off = word_attn_off + (size_t)B * NHEAD * SW * SW;

    char* p = (char*)d_ws;
    auto carve = [&](size_t bytes) -> char* {
        char* r = p; p += (bytes + 255) & ~(size_t)255; return r;
    };
    ushort* wi_bf = (ushort*)carve((size_t)B * SW * DMODEL * 2);
    ushort* wc_bf = (ushort*)carve((size_t)B * SW * DMODEL * 2);
    ushort* si_bf = (ushort*)carve((size_t)B * SS * DMODEL * 2);
    ushort* sc_bf = (ushort*)carve((size_t)B * SS * DMODEL * 2);
    ushort* wt[8];
    for (int i = 0; i < 8; ++i) wt[i] = (ushort*)carve((size_t)DMODEL * DMODEL * 2);
    ushort* Qw  = (ushort*)carve((size_t)B * SW * DMODEL * 2);
    ushort* Kw  = (ushort*)carve((size_t)B * SW * DMODEL * 2);
    ushort* VTw = (ushort*)carve((size_t)B * SW * DMODEL * 2);
    ushort* Qs  = (ushort*)carve((size_t)B * SS * DMODEL * 2);
    ushort* Ks  = (ushort*)carve((size_t)B * SS * DMODEL * 2);
    ushort* VTs = (ushort*)carve((size_t)B * SS * DMODEL * 2);
    ushort* ctxw = (ushort*)carve((size_t)B * SW * DMODEL * 2);
    ushort* ctxs = (ushort*)carve((size_t)B * SS * DMODEL * 2);

    auto cvt = [&](const float* x, ushort* y, size_t n) {
        int n4 = (int)(n / 4);
        k_f32_to_bf16<<<(n4 + 255) / 256, 256, 0, stream>>>(x, y, n4);
    };
    cvt(word_inputs,      wi_bf, (size_t)B * SW * DMODEL);
    cvt(word_context,     wc_bf, (size_t)B * SW * DMODEL);
    cvt(sentence_inputs,  si_bf, (size_t)B * SS * DMODEL);
    cvt(sentence_context, sc_bf, (size_t)B * SS * DMODEL);

    dim3 tb(32, 8), tg(DMODEL / 32, DMODEL / 32);
    for (int i = 0; i < 8; ++i)
        k_transpose_w<<<tg, tb, 0, stream>>>((const float*)d_in[6 + 2 * i], wt[i]);

    const float qscale = 0.03125f;   // 1/sqrt(1024)
    const int   lw = 10, ls = 8;     // log2(SW), log2(SS)
    dim3 gw(B * SW / 64, DMODEL / 64);
    dim3 gs(B * SS / 64, DMODEL / 64);
    k_gemm_bf16<<<gw, 32, 0, stream>>>(wi_bf, wt[0], (const float*)d_in[7],  Qw,  nullptr, lw, qscale, 0);
    k_gemm_bf16<<<gw, 32, 0, stream>>>(wc_bf, wt[1], (const float*)d_in[9],  Kw,  nullptr, lw, 1.0f,   0);
    k_gemm_bf16<<<gw, 32, 0, stream>>>(wc_bf, wt[2], (const float*)d_in[11], VTw, nullptr, lw, 1.0f,   2);
    k_gemm_bf16<<<gs, 32, 0, stream>>>(si_bf, wt[4], (const float*)d_in[15], Qs,  nullptr, ls, qscale, 0);
    k_gemm_bf16<<<gs, 32, 0, stream>>>(sc_bf, wt[5], (const float*)d_in[17], Ks,  nullptr, ls, 1.0f,   0);
    k_gemm_bf16<<<gs, 32, 0, stream>>>(sc_bf, wt[6], (const float*)d_in[19], VTs, nullptr, ls, 1.0f,   2);

    k_attn<1024><<<dim3(SW / 16, NHEAD, B), 32, 16 * (SW + 16) * 4, stream>>>(
        Qw, Kw, VTw, word_mask, out + word_attn_off, ctxw);
    k_attn<256><<<dim3(SS / 16, NHEAD, B), 32, 16 * (SS + 16) * 4, stream>>>(
        Qs, Ks, VTs, sentence_mask, out + sent_attn_off, ctxs);

    k_gemm_bf16<<<gw, 32, 0, stream>>>(ctxw, wt[3], (const float*)d_in[13], nullptr, out + word_out_off, lw, 1.0f, 1);
    k_gemm_bf16<<<gs, 32, 0, stream>>>(ctxs, wt[7], (const float*)d_in[21], nullptr, out + sent_out_off, ls, 1.0f, 1);
}
